// myGlobalAttentionGATNet3_36515811950803
// MI455X (gfx1250) — compile-verified, matrix operand order checked by reading, not run on hardware
//
#include <hip/hip_runtime.h>
#include <hip/hip_bf16.h>

// ---------------------------------------------------------------------------
// GlobalAttention GAT network for gfx1250 (MI455X), wave32 + WMMA bf16.
//
//   1) h1   = x @ W1                      (WMMA bf16, W pre-swizzled in LDS)
//   2) GAT attention layer 1 (edge softmax via float-atomic-max/add)
//   3) x2   = relu(agg1 + b1)
//   4) h2   = x2 @ W2                     (WMMA)
//   5) GAT attention layer 2
//   6) hF   = agg2 + b2
//   7) gate = hF @ gate_W + gate_b ; segment softmax over sorted batch ids
//   8) pooled[G,128] = sum attn * hF ; relu
//   9) out  = pooled @ lin_W + lin_b      (WMMA, 1 block)
// ---------------------------------------------------------------------------

#define NNODES 100000
#define NEDGES 1600000
#define F_IN   256
#define F_MID  128
#define NGRAPH 128
#define F_OUT  64
#define ORD_NEGINF 0x007FFFFFu   // f2ord(-inf)

typedef __attribute__((ext_vector_type(16))) __bf16 v16bf;
typedef __attribute__((ext_vector_type(8)))  float  v8f;

__device__ __forceinline__ unsigned f2ord(float f) {
    unsigned u = __float_as_uint(f);
    return (u & 0x80000000u) ? ~u : (u | 0x80000000u);
}
__device__ __forceinline__ float ord2f(unsigned u) {
    return (u & 0x80000000u) ? __uint_as_float(u ^ 0x80000000u)
                             : __uint_as_float(~u);
}

// ---------------------------------------------------------------------------
// Out[M, NT*16] = A[M,K] @ W[K, NT*16] (+bias, +relu as compile-time options).
//
// W is staged in LDS *pre-swizzled into the WMMA B-fragment layout*:
//   sW[((kc*NT + t)*32 + lane)*16 + j]  with
//   k   = kc*32 + (lane>>4)*8 + (j<8 ? j : 16 + j-8)
//   col = t*16 + (lane&15)
// so each lane's 16-element fragment is one contiguous 32-byte LDS read
// (2x ds_load_b128, no v_mov_b16 repacking).
//
// 8 waves/block; each wave owns 16 rows and all NT column tiles; K stepped
// 32 at a time via v_wmma_f32_16x16x32_bf16 (f32 accumulate). A stream is
// read as float4 pairs (global_load_b128) + v_cvt_pk_bf16_f32, with a
// global_prefetch of the next K-chunk.
// ---------------------------------------------------------------------------
template <int K, int NT, bool BIAS, bool RELU>
__global__ __launch_bounds__(256) void gemm_wmma_bf16(
    const float* __restrict__ A, const float* __restrict__ W,
    const float* __restrict__ bias, float* __restrict__ Out,
    int M)
{
    constexpr int NOUT = NT * 16;
    constexpr int NKC  = K / 32;
    __shared__ __align__(16) __bf16 sW[NKC * NT * 32 * 16];

    // cooperative swizzled fill of W
    for (int idx = threadIdx.x; idx < NKC * NT * 32 * 16; idx += 256) {
        int j    = idx & 15;
        int ln   = (idx >> 4) & 31;
        int tk   = idx >> 9;          // kc*NT + t
        int t    = tk % NT;
        int kc   = tk / NT;
        int k    = kc * 32 + ((ln >> 4) << 3) + ((j < 8) ? j : (8 + j));
        int col  = t * 16 + (ln & 15);
        sW[idx]  = (__bf16)W[(size_t)k * NOUT + col];
    }
    __syncthreads();

    const int wave = threadIdx.x >> 5;
    const int lane = threadIdx.x & 31;
    const int row0 = blockIdx.x * 128 + wave * 16;
    if (row0 >= M) return;                 // wave-uniform: EXEC stays all-1s

    const int mrow  = lane & 15;
    const int khalf = (lane >> 4) * 8;

    v8f acc[NT];
    const v8f vzero = {0.f, 0.f, 0.f, 0.f, 0.f, 0.f, 0.f, 0.f};
    #pragma unroll
    for (int t = 0; t < NT; ++t) acc[t] = vzero;

    int arow = row0 + mrow;
    if (arow >= M) arow = M - 1;           // tail: duplicate read (safe)
    const float* arowp = A + (size_t)arow * K;

    for (int kc = 0; kc < K; kc += 32) {
        const float* ap = arowp + kc + khalf;
        if (kc + 32 < K) __builtin_prefetch(ap + 32, 0, 0);

        float4 a0 = *(const float4*)(ap + 0);
        float4 a1 = *(const float4*)(ap + 4);
        float4 a2 = *(const float4*)(ap + 16);
        float4 a3 = *(const float4*)(ap + 20);
        v16bf afrag;
        afrag[0]  = (__bf16)a0.x;  afrag[1]  = (__bf16)a0.y;
        afrag[2]  = (__bf16)a0.z;  afrag[3]  = (__bf16)a0.w;
        afrag[4]  = (__bf16)a1.x;  afrag[5]  = (__bf16)a1.y;
        afrag[6]  = (__bf16)a1.z;  afrag[7]  = (__bf16)a1.w;
        afrag[8]  = (__bf16)a2.x;  afrag[9]  = (__bf16)a2.y;
        afrag[10] = (__bf16)a2.z;  afrag[11] = (__bf16)a2.w;
        afrag[12] = (__bf16)a3.x;  afrag[13] = (__bf16)a3.y;
        afrag[14] = (__bf16)a3.z;  afrag[15] = (__bf16)a3.w;

        const __bf16* bbase = sW + (size_t)((kc >> 5) * NT) * 512 + lane * 16;
        #pragma unroll
        for (int t = 0; t < NT; ++t) {
            v16bf bfrag = *(const v16bf*)(bbase + (size_t)t * 512);
            acc[t] = __builtin_amdgcn_wmma_f32_16x16x32_bf16(
                false, afrag, false, bfrag, (short)0, acc[t], false, false);
        }
    }

    const int  mout = (lane >> 4) * 8;     // C/D: upper half-wave = rows +8
    const int  ncol = lane & 15;
    const bool full = (row0 + 16 <= M);
    #pragma unroll
    for (int t = 0; t < NT; ++t) {
        const int col = t * 16 + ncol;
        float bv = 0.f;
        if (BIAS) bv = bias[col];
        float* op = Out + (size_t)(row0 + mout) * NOUT + col;
        if (full) {
            #pragma unroll
            for (int r = 0; r < 8; ++r) {
                float v = acc[t][r];
                if (BIAS) v += bv;
                if (RELU) v = v > 0.f ? v : 0.f;
                op[(size_t)r * NOUT] = v;
            }
        } else {
            #pragma unroll
            for (int r = 0; r < 8; ++r) {
                if (row0 + mout + r < M) {
                    float v = acc[t][r];
                    if (BIAS) v += bv;
                    if (RELU) v = v > 0.f ? v : 0.f;
                    op[(size_t)r * NOUT] = v;
                }
            }
        }
    }
}

// ---------------------------------------------------------------------------
__global__ void fill_u32(unsigned* __restrict__ p, unsigned v, int n) {
    int i = blockIdx.x * blockDim.x + threadIdx.x;
    if (i < n) p[i] = v;
}

__global__ void add_bias_act(const float* __restrict__ in,
                             const float* __restrict__ b,
                             float* __restrict__ out, int n, int relu) {
    int i = blockIdx.x * blockDim.x + threadIdx.x;
    if (i >= n) return;
    float v = in[i] + b[i & (F_MID - 1)];
    if (relu) v = v > 0.f ? v : 0.f;
    out[i] = v;
}

__global__ void relu_inplace(float* __restrict__ p, int n) {
    int i = blockIdx.x * blockDim.x + threadIdx.x;
    if (i < n) { float v = p[i]; p[i] = v > 0.f ? v : 0.f; }
}

// ---------------------------------------------------------------------------
// Per-node dots: as[n]=h[n]·a_src, ad[n]=h[n]·a_dst. Wave32 per node,
// float4 per lane (one global_load_b128), shfl_xor reduction.
// ---------------------------------------------------------------------------
__global__ __launch_bounds__(256) void node_dots(
    const float* __restrict__ h, const float* __restrict__ asrc,
    const float* __restrict__ adst, float* __restrict__ out_s,
    float* __restrict__ out_d, int n)
{
    int wid  = (blockIdx.x * 256 + threadIdx.x) >> 5;
    int lane = threadIdx.x & 31;
    if (wid >= n) return;
    const float* hp = h + (size_t)wid * F_MID + lane * 4;
    float4 v  = *(const float4*)hp;
    float4 vs = *(const float4*)(asrc + lane * 4);
    float4 vd = *(const float4*)(adst + lane * 4);
    float ss = v.x * vs.x + v.y * vs.y + v.z * vs.z + v.w * vs.w;
    float sd = v.x * vd.x + v.y * vd.y + v.z * vd.z + v.w * vd.w;
    #pragma unroll
    for (int off = 16; off > 0; off >>= 1) {
        ss += __shfl_xor(ss, off, 32);
        sd += __shfl_xor(sd, off, 32);
    }
    if (lane == 0) { out_s[wid] = ss; out_d[wid] = sd; }
}

__global__ __launch_bounds__(256) void gate_dot(
    const float* __restrict__ h, const float* __restrict__ gw,
    const float* __restrict__ gb, float* __restrict__ gate, int n)
{
    int wid  = (blockIdx.x * 256 + threadIdx.x) >> 5;
    int lane = threadIdx.x & 31;
    if (wid >= n) return;
    float4 v  = *(const float4*)(h + (size_t)wid * F_MID + lane * 4);
    float4 w  = *(const float4*)(gw + lane * 4);
    float acc = v.x * w.x + v.y * w.y + v.z * w.z + v.w * w.w;
    #pragma unroll
    for (int off = 16; off > 0; off >>= 1) acc += __shfl_xor(acc, off, 32);
    if (lane == 0) gate[wid] = acc + gb[0];
}

// ---------------------------------------------------------------------------
// Edge passes over E real edges + N appended self-loops.
// ---------------------------------------------------------------------------
__device__ __forceinline__ void edge_sd(const int* __restrict__ ei, int e,
                                        int& s, int& d) {
    if (e < NEDGES) { s = ei[e]; d = ei[NEDGES + e]; }
    else            { s = d = e - NEDGES; }
}

__global__ void edge_max(const int* __restrict__ ei,
                         const float* __restrict__ as,
                         const float* __restrict__ ad,
                         unsigned* __restrict__ m, int tot)
{
    int e = blockIdx.x * blockDim.x + threadIdx.x;
    if (e >= tot) return;
    int s, d; edge_sd(ei, e, s, d);
    float al = as[s] + ad[d];
    al = al > 0.f ? al : 0.2f * al;          // leaky_relu(0.2)
    atomicMax(&m[d], f2ord(al));
}

__global__ void edge_expsum(const int* __restrict__ ei,
                            const float* __restrict__ as,
                            const float* __restrict__ ad,
                            const unsigned* __restrict__ m,
                            float* __restrict__ ew,
                            float* __restrict__ ssum, int tot)
{
    int e = blockIdx.x * blockDim.x + threadIdx.x;
    if (e >= tot) return;
    int s, d; edge_sd(ei, e, s, d);
    float al = as[s] + ad[d];
    al = al > 0.f ? al : 0.2f * al;
    float ex = __expf(al - ord2f(m[d]));
    ew[e] = ex;
    atomicAdd(&ssum[d], ex);
}

// wave32 per edge: one float4 gather per lane (512B/edge coalesced), then
// scatter-add via global_atomic_add_f32 into agg[dst].
__global__ __launch_bounds__(256) void edge_aggregate(
    const int* __restrict__ ei, const float* __restrict__ ew,
    const float* __restrict__ ssum, const float* __restrict__ h,
    float* __restrict__ agg, int tot)
{
    int wid  = (blockIdx.x * 256 + threadIdx.x) >> 5;
    int lane = threadIdx.x & 31;
    if (wid >= tot) return;
    int s, d; edge_sd(ei, wid, s, d);
    float coef = ew[wid] / (ssum[d] + 1e-16f);
    float4 v = *(const float4*)(h + (size_t)s * F_MID + lane * 4);
    float* op = agg + (size_t)d * F_MID + lane * 4;
    atomicAdd(op + 0, v.x * coef);
    atomicAdd(op + 1, v.y * coef);
    atomicAdd(op + 2, v.z * coef);
    atomicAdd(op + 3, v.w * coef);
}

// ---------------------------------------------------------------------------
// Global-attention pooling over sorted batch ids
// ---------------------------------------------------------------------------
__global__ void pool_max(const int* __restrict__ batch,
                         const float* __restrict__ gate,
                         unsigned* __restrict__ gm, int n)
{
    int i = blockIdx.x * blockDim.x + threadIdx.x;
    if (i < n) atomicMax(&gm[batch[i]], f2ord(gate[i]));
}

__global__ void pool_expsum(const int* __restrict__ batch,
                            const float* __restrict__ gate,
                            const unsigned* __restrict__ gm,
                            float* __restrict__ ge,
                            float* __restrict__ gs, int n)
{
    int i = blockIdx.x * blockDim.x + threadIdx.x;
    if (i >= n) return;
    int b = batch[i];
    float ex = __expf(gate[i] - ord2f(gm[b]));
    ge[i] = ex;
    atomicAdd(&gs[b], ex);
}

__global__ __launch_bounds__(256) void pool_acc(
    const int* __restrict__ batch, const float* __restrict__ ge,
    const float* __restrict__ gs, const float* __restrict__ h,
    float* __restrict__ pooled, int n)
{
    int wid  = (blockIdx.x * 256 + threadIdx.x) >> 5;
    int lane = threadIdx.x & 31;
    if (wid >= n) return;
    int b = batch[wid];
    float coef = ge[wid] / (gs[b] + 1e-16f);
    float4 v = *(const float4*)(h + (size_t)wid * F_MID + lane * 4);
    float* pp = pooled + (size_t)b * F_MID + lane * 4;
    atomicAdd(pp + 0, v.x * coef);
    atomicAdd(pp + 1, v.y * coef);
    atomicAdd(pp + 2, v.z * coef);
    atomicAdd(pp + 3, v.w * coef);
}

// ---------------------------------------------------------------------------
extern "C" void kernel_launch(void* const* d_in, const int* in_sizes, int n_in,
                              void* d_out, int out_size, void* d_ws, size_t ws_size,
                              hipStream_t stream)
{
    (void)in_sizes; (void)n_in; (void)out_size; (void)ws_size;

    const float* x     = (const float*)d_in[0];
    const int*   ei    = (const int*)  d_in[1];   // [2,E] row-major
    const int*   batch = (const int*)  d_in[2];
    const float* W1    = (const float*)d_in[3];
    const float* a1s   = (const float*)d_in[4];
    const float* a1d   = (const float*)d_in[5];
    const float* b1    = (const float*)d_in[6];
    const float* W2    = (const float*)d_in[7];
    const float* a2s   = (const float*)d_in[8];
    const float* a2d   = (const float*)d_in[9];
    const float* b2    = (const float*)d_in[10];
    const float* gateW = (const float*)d_in[11];
    const float* gateB = (const float*)d_in[12];
    const float* linW  = (const float*)d_in[13];
    const float* linB  = (const float*)d_in[14];
    float* out         = (float*)d_out;

    const int N   = NNODES;
    const int TOT = NEDGES + NNODES;

    // ---- workspace carve-up (~162 MB; mostly resident in 192 MB L2) ----
    float*    bufA   = (float*)d_ws;               // [N,128] h1 / h2
    float*    bufB   = bufA + (size_t)N * F_MID;   // [N,128] aggregation
    float*    bufC   = bufB + (size_t)N * F_MID;   // [N,128] x2 / h_final
    float*    as_    = bufC + (size_t)N * F_MID;   // [N]
    float*    ad_    = as_ + N;                    // [N]
    unsigned* m_     = (unsigned*)(ad_ + N);       // [N]
    float*    s_     = (float*)(m_ + N);           // [N]
    float*    ew_    = s_ + N;                     // [TOT] per-edge/per-node exp
    float*    gate_  = ew_ + TOT;                  // [N]
    unsigned* gm_    = (unsigned*)(gate_ + N);     // [G]
    float*    gs_    = (float*)(gm_ + NGRAPH);     // [G]
    float*    pooled = gs_ + NGRAPH;               // [G,128]

    const int BLK       = 256;
    const int gRows     = (N + 127) / 128;
    const int gN        = (N + BLK - 1) / BLK;
    const int gNF       = (N * F_MID + BLK - 1) / BLK;
    const int gEdge     = (TOT + BLK - 1) / BLK;
    const int gEdgeWave = (TOT + 7) / 8;
    const int gNodeWave = (N + 7) / 8;
    const int gPool     = (NGRAPH * F_MID + BLK - 1) / BLK;

    // ================= Layer 1 =================
    gemm_wmma_bf16<F_IN, F_MID / 16, false, false><<<gRows, BLK, 0, stream>>>(
        x, W1, nullptr, bufA, N);
    node_dots<<<gNodeWave, BLK, 0, stream>>>(bufA, a1s, a1d, as_, ad_, N);

    fill_u32<<<gNF, BLK, 0, stream>>>((unsigned*)bufB, 0u, N * F_MID);
    fill_u32<<<gN, BLK, 0, stream>>>(m_, ORD_NEGINF, N);
    fill_u32<<<gN, BLK, 0, stream>>>((unsigned*)s_, 0u, N);

    edge_max<<<gEdge, BLK, 0, stream>>>(ei, as_, ad_, m_, TOT);
    edge_expsum<<<gEdge, BLK, 0, stream>>>(ei, as_, ad_, m_, ew_, s_, TOT);
    edge_aggregate<<<gEdgeWave, BLK, 0, stream>>>(ei, ew_, s_, bufA, bufB, TOT);
    add_bias_act<<<gNF, BLK, 0, stream>>>(bufB, b1, bufC, N * F_MID, 1);

    // ================= Layer 2 =================
    gemm_wmma_bf16<F_MID, F_MID / 16, false, false><<<gRows, BLK, 0, stream>>>(
        bufC, W2, nullptr, bufA, N);
    node_dots<<<gNodeWave, BLK, 0, stream>>>(bufA, a2s, a2d, as_, ad_, N);

    fill_u32<<<gNF, BLK, 0, stream>>>((unsigned*)bufB, 0u, N * F_MID);
    fill_u32<<<gN, BLK, 0, stream>>>(m_, ORD_NEGINF, N);
    fill_u32<<<gN, BLK, 0, stream>>>((unsigned*)s_, 0u, N);

    edge_max<<<gEdge, BLK, 0, stream>>>(ei, as_, ad_, m_, TOT);
    edge_expsum<<<gEdge, BLK, 0, stream>>>(ei, as_, ad_, m_, ew_, s_, TOT);
    edge_aggregate<<<gEdgeWave, BLK, 0, stream>>>(ei, ew_, s_, bufA, bufB, TOT);
    add_bias_act<<<gNF, BLK, 0, stream>>>(bufB, b2, bufC, N * F_MID, 0);

    // ================= GlobalAttention pooling =================
    gate_dot<<<gNodeWave, BLK, 0, stream>>>(bufC, gateW, gateB, gate_, N);

    fill_u32<<<1, BLK, 0, stream>>>(gm_, ORD_NEGINF, NGRAPH);
    fill_u32<<<1, BLK, 0, stream>>>((unsigned*)gs_, 0u, NGRAPH);
    fill_u32<<<gPool, BLK, 0, stream>>>((unsigned*)pooled, 0u, NGRAPH * F_MID);

    pool_max<<<gN, BLK, 0, stream>>>(batch, gate_, gm_, N);
    pool_expsum<<<gN, BLK, 0, stream>>>(batch, gate_, gm_, ew_, gs_, N); // ew_ reused: per-node exp
    pool_acc<<<gNodeWave, BLK, 0, stream>>>(batch, ew_, gs_, bufC, pooled, N);
    relu_inplace<<<gPool, BLK, 0, stream>>>(pooled, NGRAPH * F_MID);

    // ================= Final linear =================
    gemm_wmma_bf16<F_MID, F_OUT / 16, true, false><<<1, BLK, 0, stream>>>(
        pooled, linW, linB, out, NGRAPH);
}
